// MoeLayer_2559800509230
// MI455X (gfx1250) — matrix operand order repaired and checked by reading, not compiled, not run on hardware
//
#include <hip/hip_runtime.h>
#include <hip/hip_bf16.h>

// ---------------------------------------------------------------------------
// MoE layer for MI455X (gfx1250, wave32, WMMA).
//   T=32768 tokens, C=512, E=8 experts, H=16, K(top)=2.
// GEMMs run as v_wmma_f32_16x16x32_bf16 (f32 accumulate).
// ---------------------------------------------------------------------------

#define TT   32768          // tokens (B*N)
#define CC   512            // hidden dim
#define EE   8              // experts
#define HH   16             // expert hidden
#define EH   128            // E*H
#define NN   4096           // sequence length
#define BN_EPS 1e-5f

typedef __attribute__((ext_vector_type(16))) __bf16 v16bf;
typedef __attribute__((ext_vector_type(8)))  float  v8f;

struct alignas(32) Bf16Frag { uint4 lo; uint4 hi; };
struct alignas(16) BF8      { __bf16 v[8]; };

// A-operand fragment: 8 bf16 at p[0..7] and 8 bf16 at p[16..23]
static __device__ __forceinline__ v16bf ldA(const __bf16* p) {
    Bf16Frag f;
    f.lo = *reinterpret_cast<const uint4*>(p);
    f.hi = *reinterpret_cast<const uint4*>(p + 16);
    return __builtin_bit_cast(v16bf, f);
}
// B-operand fragment: 8 bf16 at p[0..7] and 8 bf16 at p[16..23]
static __device__ __forceinline__ v16bf ldB(const __bf16* p) {
    Bf16Frag f;
    f.lo = *reinterpret_cast<const uint4*>(p);
    f.hi = *reinterpret_cast<const uint4*>(p + 16);
    return __builtin_bit_cast(v16bf, f);
}

// --------------------------- prep kernels ----------------------------------

__global__ void k_zero(float* p, int n) {
    for (int i = blockIdx.x * blockDim.x + threadIdx.x; i < n;
         i += gridDim.x * blockDim.x)
        p[i] = 0.f;
}

__global__ void k_f2b(const float* __restrict__ in, __bf16* __restrict__ ob, int n) {
    int i = blockIdx.x * blockDim.x + threadIdx.x;
    if (i < n) ob[i] = (__bf16)in[i];
}

// w1  [E, C, H] -> w1t [EH, C]   (row j=e*16+h contiguous along c)
__global__ void k_prep_w1t(const float* __restrict__ w1, __bf16* __restrict__ w1t) {
    int i = blockIdx.x * blockDim.x + threadIdx.x;   // 65536
    if (i < EH * CC) {
        int j = i >> 9, c = i & (CC - 1);
        int e = j >> 4, h = j & 15;
        w1t[i] = (__bf16)w1[e * (CC * HH) + c * HH + h];
    }
}

// w2  [E, H, C] -> w2t [C, EH]   (row c contiguous along j=e*16+h)
__global__ void k_prep_w2t(const float* __restrict__ w2, __bf16* __restrict__ w2t) {
    int i = blockIdx.x * blockDim.x + threadIdx.x;   // 65536
    if (i < CC * EH) {
        int c = i >> 7, j = i & (EH - 1);
        int e = j >> 4, h = j & 15;
        w2t[i] = (__bf16)w2[e * (HH * CC) + h * CC + c];
    }
}

// ------------------- gate GEMM: Ht = (X @ Wg1^T + bg1), BN partials --------
// Block: 256 thr = 8 waves; wave tile 16(M) x 64(N); block tile 32 x 256.
// Grid (TT/32, CC/256). Ht stored channel-major: Ht[c*TT + t].
__global__ __launch_bounds__(256) void k_gate(
    const __bf16* __restrict__ xb, const __bf16* __restrict__ wg1b,
    const float* __restrict__ bg1, __bf16* __restrict__ Ht,
    float* __restrict__ wsum, float* __restrict__ wsq)
{
    const int lane = threadIdx.x & 31;
    const int w    = threadIdx.x >> 5;
    const int mt   = blockIdx.x * 32 + (w & 1) * 16;
    const int nb   = blockIdx.y * 256 + (w >> 1) * 64;
    const int mrow = mt + (lane & 15);
    const int aoff = ((lane >> 4) & 1) * 8;
    const int boff = ((lane >> 4) & 1) * 8;

    v8f acc[4] = {};
    for (int kt = 0; kt < CC / 32; ++kt) {
        const int k = kt * 32;
        v16bf a = ldA(xb + (size_t)mrow * CC + k + aoff);
#pragma unroll
        for (int j = 0; j < 4; ++j) {
            const int n = nb + j * 16 + (lane & 15);
            v16bf b = ldB(wg1b + (size_t)n * CC + k + boff);
            acc[j] = __builtin_amdgcn_wmma_f32_16x16x32_bf16(
                false, a, false, b, (short)0, acc[j], false, false);
        }
    }

    const int roff = mt + ((lane >> 4) << 3);   // first of 8 consecutive tokens
#pragma unroll
    for (int j = 0; j < 4; ++j) {
        const int col  = nb + j * 16 + (lane & 15);
        const float bc = bg1[col];
        float s = 0.f, sq = 0.f;
        BF8 pk;
#pragma unroll
        for (int v = 0; v < 8; ++v) {
            float val = acc[j][v] + bc;
            s  += val;
            sq += val * val;
            pk.v[v] = (__bf16)val;
        }
        *reinterpret_cast<uint4*>(Ht + (size_t)col * TT + roff) =
            __builtin_bit_cast(uint4, pk);
        atomicAdd(&wsum[col], s);
        atomicAdd(&wsq[col], sq);
    }
}

// ------------------------------- BN finalize -------------------------------
__global__ void k_bnstats(const float* __restrict__ wsum, const float* __restrict__ wsq,
                          float* __restrict__ wmu, float* __restrict__ wrs)
{
    int c = blockIdx.x * blockDim.x + threadIdx.x;
    if (c < CC) {
        float m = wsum[c] * (1.f / TT);
        float v = wsq[c] * (1.f / TT) - m * m;
        wmu[c] = m;
        wrs[c] = rsqrtf(v + BN_EPS);
    }
}

// ---------------- router: BN -> relu -> logits -> top2 softmax -------------
__global__ __launch_bounds__(256) void k_router(
    const __bf16* __restrict__ Ht, const float* __restrict__ wmu,
    const float* __restrict__ wrs, const float* __restrict__ gamma,
    const float* __restrict__ beta, const float* __restrict__ wg2,
    const float* __restrict__ bg2, float* __restrict__ wts,
    float* __restrict__ usage)
{
    __shared__ float sw[EE * CC];
    __shared__ float su[EE];
    const int tid = threadIdx.x;
    for (int i = tid; i < EE * CC; i += 256) sw[i] = wg2[i];
    if (tid < EE) su[tid] = 0.f;
    __syncthreads();

    const int t = blockIdx.x * 256 + tid;
    float acc[EE];
#pragma unroll
    for (int e = 0; e < EE; ++e) acc[e] = bg2[e];

    for (int c = 0; c < CC; ++c) {
        float hv = (float)Ht[(size_t)c * TT + t];
        float hn = gamma[c] * (hv - wmu[c]) * wrs[c] + beta[c];
        hn = hn > 0.f ? hn : 0.f;
#pragma unroll
        for (int e = 0; e < EE; ++e) acc[e] += hn * sw[e * CC + c];
    }

    int i1 = 0; float v1 = acc[0];
#pragma unroll
    for (int e = 1; e < EE; ++e) if (acc[e] > v1) { v1 = acc[e]; i1 = e; }
    int i2 = -1; float v2 = -3.4e38f;
#pragma unroll
    for (int e = 0; e < EE; ++e)
        if (e != i1 && acc[e] > v2) { v2 = acc[e]; i2 = e; }

    float e2 = __expf(v2 - v1);
    float wA = 1.f / (1.f + e2);
    float wB = e2 * wA;
#pragma unroll
    for (int e = 0; e < EE; ++e)
        wts[(size_t)t * EE + e] = (e == i1) ? wA : ((e == i2) ? wB : 0.f);

    atomicAdd(&su[i1], wA);
    atomicAdd(&su[i2], wB);
    __syncthreads();
    if (tid < EE) atomicAdd(&usage[tid], su[tid]);
}

// ----- fused experts: H1=X@W1' ; G=w*relu(H1+b1) ; OUT=G@W2' (+bias mix) ---
// Block 256 thr = 8 waves; each wave owns a 16-token tile. LDS transposes the
// [16x128] intermediate from D-layout (GEMM1) to A-layout (GEMM2).
__global__ __launch_bounds__(256) void k_expert(
    const __bf16* __restrict__ xb, const __bf16* __restrict__ w1t,
    const __bf16* __restrict__ w2t, const float* __restrict__ b1,
    const float* __restrict__ b2, const float* __restrict__ wts,
    float* __restrict__ out)
{
    __shared__ __bf16 gbuf[8][16 * EH];
    __shared__ float  wbuf[8][16 * EE];

    const int lane = threadIdx.x & 31;
    const int w    = threadIdx.x >> 5;
    const int t0   = (blockIdx.x * 8 + w) * 16;
    const int m    = lane & 15;
    const int aoff = ((lane >> 4) & 1) * 8;
    const int boff = ((lane >> 4) & 1) * 8;
    const int m0   = (lane >> 4) << 3;

#pragma unroll
    for (int i = 0; i < 4; ++i)
        wbuf[w][lane * 4 + i] = wts[(size_t)t0 * EE + lane * 4 + i];
    __syncthreads();

    // ---- stage 1: [16 x 512] @ [512 x 128] ----
    v8f acc[8] = {};
    for (int kt = 0; kt < CC / 32; ++kt) {
        const int k = kt * 32;
        v16bf a = ldA(xb + (size_t)(t0 + m) * CC + k + aoff);
#pragma unroll
        for (int j = 0; j < 8; ++j) {
            const int n = j * 16 + m;
            v16bf b = ldB(w1t + (size_t)n * CC + k + boff);
            acc[j] = __builtin_amdgcn_wmma_f32_16x16x32_bf16(
                false, a, false, b, (short)0, acc[j], false, false);
        }
    }

    // epilogue: + b1, relu, * gate weight, -> LDS (row-major [16][128] bf16)
#pragma unroll
    for (int j = 0; j < 8; ++j) {
        const int col = j * 16 + m;          // col = e*16 + h
        const int e   = col >> 4;
        const float bb = b1[col];
#pragma unroll
        for (int v = 0; v < 8; ++v) {
            const int row = m0 + v;
            float val = acc[j][v] + bb;
            val = val > 0.f ? val : 0.f;
            val *= wbuf[w][row * EE + e];
            gbuf[w][row * EH + col] = (__bf16)val;
        }
    }
    __syncthreads();

    // hoist gate weights for this lane's 8 rows
    float wreg[8][EE];
#pragma unroll
    for (int v = 0; v < 8; ++v)
#pragma unroll
        for (int e = 0; e < EE; ++e)
            wreg[v][e] = wbuf[w][(m0 + v) * EE + e];

    // preload stage-2 A fragments from LDS
    v16bf a2[4];
#pragma unroll
    for (int kt = 0; kt < 4; ++kt)
        a2[kt] = ldA(&gbuf[w][m * EH + kt * 32 + aoff]);

    // ---- stage 2: [16 x 128] @ [128 x 512], write out[B,C,N] transposed ----
    for (int n2 = 0; n2 < CC / 16; ++n2) {
        v8f acc2 = {};
#pragma unroll
        for (int kt = 0; kt < 4; ++kt) {
            const int n = n2 * 16 + m;
            v16bf b = ldB(w2t + (size_t)n * EH + kt * 32 + boff);
            acc2 = __builtin_amdgcn_wmma_f32_16x16x32_bf16(
                false, a2[kt], false, b, (short)0, acc2, false, false);
        }
        const int c = n2 * 16 + m;
        float b2e[EE];
#pragma unroll
        for (int e = 0; e < EE; ++e) b2e[e] = b2[e * CC + c];

        float vals[8];
#pragma unroll
        for (int v = 0; v < 8; ++v) {
            float bb = 0.f;
#pragma unroll
            for (int e = 0; e < EE; ++e) bb += wreg[v][e] * b2e[e];
            vals[v] = acc2[v] + bb;
        }
        // token t0+m0+v are consecutive -> contiguous along N of out[B,C,N]
        const int t    = t0 + m0;
        const int bidx = t >> 12;           // / NN
        const int nn   = t & (NN - 1);
        float* op = out + ((size_t)(bidx * CC + c) * NN) + nn;
        *reinterpret_cast<float4*>(op)     = make_float4(vals[0], vals[1], vals[2], vals[3]);
        *reinterpret_cast<float4*>(op + 4) = make_float4(vals[4], vals[5], vals[6], vals[7]);
    }
}

// ------------------------------- lb loss -----------------------------------
__global__ void k_loss(const float* __restrict__ usage, float* __restrict__ out) {
    if (threadIdx.x == 0 && blockIdx.x == 0) {
        float lb = 0.f;
#pragma unroll
        for (int e = 0; e < EE; ++e) {
            float u = usage[e] * (1.f / TT);
            lb += u * u;
        }
        out[(size_t)8 * CC * NN] = lb * (float)EE;
    }
}

// ---------------------------------------------------------------------------

extern "C" void kernel_launch(void* const* d_in, const int* in_sizes, int n_in,
                              void* d_out, int out_size, void* d_ws, size_t ws_size,
                              hipStream_t stream) {
    (void)in_sizes; (void)n_in; (void)out_size; (void)ws_size;
    const float* x     = (const float*)d_in[0];
    const float* wg1   = (const float*)d_in[1];
    const float* bg1   = (const float*)d_in[2];
    const float* gamma = (const float*)d_in[3];
    const float* beta  = (const float*)d_in[4];
    const float* wg2   = (const float*)d_in[5];
    const float* bg2   = (const float*)d_in[6];
    const float* w1    = (const float*)d_in[7];
    const float* b1    = (const float*)d_in[8];
    const float* w2    = (const float*)d_in[9];
    const float* b2    = (const float*)d_in[10];
    float* out = (float*)d_out;

    // workspace carve-up (~67 MB)
    __bf16* Ht   = (__bf16*)d_ws;                 // [CC][TT]      16.7M bf16
    __bf16* xb   = Ht   + (size_t)TT * CC;        // [TT][CC]      16.7M bf16
    __bf16* wg1b = xb   + (size_t)TT * CC;        // [CC][CC]
    __bf16* w1t  = wg1b + (size_t)CC * CC;        // [EH][CC]
    __bf16* w2t  = w1t  + (size_t)EH * CC;        // [CC][EH]
    float*  wts  = (float*)(w2t + (size_t)CC * EH); // [TT][EE]
    float*  wsum = wts  + (size_t)TT * EE;        // [CC]
    float*  wsq  = wsum + CC;
    float*  wmu  = wsq  + CC;
    float*  wrs  = wmu  + CC;
    float*  wus  = wrs  + CC;                     // [EE]

    // zero accumulators (sum/sq/mu/rs/usage block)
    k_zero<<<4, 256, 0, stream>>>(wsum, 4 * CC + EE);

    // bf16 staging
    k_f2b<<<(TT * CC + 255) / 256, 256, 0, stream>>>(x, xb, TT * CC);
    k_f2b<<<(CC * CC + 255) / 256, 256, 0, stream>>>(wg1, wg1b, CC * CC);
    k_prep_w1t<<<(EH * CC + 255) / 256, 256, 0, stream>>>(w1, w1t);
    k_prep_w2t<<<(CC * EH + 255) / 256, 256, 0, stream>>>(w2, w2t);

    // gate GEMM + BN partial sums
    k_gate<<<dim3(TT / 32, CC / 256), 256, 0, stream>>>(xb, wg1b, bg1, Ht, wsum, wsq);

    // BN finalize
    k_bnstats<<<2, 256, 0, stream>>>(wsum, wsq, wmu, wrs);

    // router: logits, top-2, softmax weights, usage
    k_router<<<TT / 256, 256, 0, stream>>>(Ht, wmu, wrs, gamma, beta, wg2, bg2, wts, wus);

    // fused expert double-GEMM, writes out[B,C,N]
    k_expert<<<TT / 128, 256, 0, stream>>>(xb, w1t, w2t, b1, b2, wts, out);

    // load-balance loss scalar
    k_loss<<<1, 32, 0, stream>>>(wus, out);
}
